// Attention_257698038328
// MI455X (gfx1250) — compile-verified, matrix operand order checked by reading, not run on hardware
//
#include <hip/hip_runtime.h>
#include <hip/hip_bf16.h>

typedef __attribute__((ext_vector_type(16))) __bf16 v16bf;
typedef __attribute__((ext_vector_type(8)))  float  v8f;
typedef __attribute__((ext_vector_type(4)))  unsigned int v4u;
typedef __attribute__((ext_vector_type(8)))  int    v8i_t;
typedef __attribute__((ext_vector_type(4)))  int    v4i_t;

#define NEGF (-3.40282347e38f)

__device__ __forceinline__ v8f v8f_zero() {
    v8f z;
#pragma unroll
    for (int i = 0; i < 8; ++i) z[i] = 0.f;
    return z;
}

// ---------------------------------------------------------------------------
// TDM: async 2D tile load global -> LDS (Tensor Data Mover, gfx1250).
// D# per CDNA5 ISA ch.8: group0 = {count/lds_addr/global_addr/type},
// group1 = {data_size, pad cfg, tensor/tile dims, row stride}.
// pad_interval code: 2^(c+1) DWORDs; pad_amount code: (c+1) DWORDs.
// 6-arg builtin variant (clang-23 / therock headers).
// ---------------------------------------------------------------------------
__device__ __forceinline__ void tdm_load_2d(const void* gptr, unsigned lds_off,
                                            unsigned rows, unsigned row_len,
                                            unsigned long long row_stride,
                                            unsigned pad_int_code,
                                            unsigned pad_amt_code) {
    unsigned long long ga = (unsigned long long)gptr;
    v4u g0;
    g0[0] = 1u;                                   // count=1 (user descriptor)
    g0[1] = lds_off;                              // LDS byte address
    g0[2] = (unsigned)(ga & 0xFFFFFFFFu);         // global_addr[31:0]
    g0[3] = (unsigned)((ga >> 32) & 0x1FFFFFFu)   // global_addr[56:32]
          | 0x80000000u;                          // type=2 ("image")
    v8i_t g1;
    g1[0] = (int)((1u << 16)                      // data_size = 2 bytes
                | (1u << 20)                      // pad_enable
                | (pad_int_code << 22)
                | (pad_amt_code << 25));
    g1[1] = (int)((row_len & 0xFFFFu) << 16);     // tensor_dim0[15:0]
    g1[2] = (int)(((row_len >> 16) & 0xFFFFu)     // tensor_dim0[31:16]
                | ((rows & 0xFFFFu) << 16));      // tensor_dim1[15:0]
    g1[3] = (int)(((rows >> 16) & 0xFFFFu)        // tensor_dim1[31:16]
                | ((row_len & 0xFFFFu) << 16));   // tile_dim0
    g1[4] = (int)(rows & 0xFFFFu);                // tile_dim1 (tile_dim2 = 0)
    g1[5] = (int)(row_stride & 0xFFFFFFFFull);    // tensor_dim0_stride[31:0]
    g1[6] = (int)((row_stride >> 32) & 0xFFFFull);
    g1[7] = 0;
    v4i_t gz4 = {0, 0, 0, 0};
    v8i_t gz8;
#pragma unroll
    for (int i = 0; i < 8; ++i) gz8[i] = 0;
    __builtin_amdgcn_tensor_load_to_lds(g0, g1, gz4, gz4, gz8, 0);
}

__device__ __forceinline__ unsigned lds_offset_of(const void* p) {
    return (unsigned)(unsigned long long)(size_t)p;   // flat LDS addr low 32 = offset
}

// ---------------------------------------------------------------------------
// fp32 -> bf16 conversion
// ---------------------------------------------------------------------------
__global__ void f32_to_bf16_kernel(const float* __restrict__ in,
                                   __bf16* __restrict__ out, size_t n) {
    size_t i = (size_t)blockIdx.x * blockDim.x + threadIdx.x;
    if (i < n) out[i] = (__bf16)in[i];
}

// ---------------------------------------------------------------------------
// bf16 GEMM: C[M,N] = A[M,K] @ B[K,N]; block tile 128x128, K-chunk 32.
// 256 threads = 8 waves in a 4x2 grid; each wave -> 32x64 (8 wmma / chunk).
// TDM double-buffered LDS staging, overlapped with compute.
// ---------------------------------------------------------------------------
#define GA_LD 40    // 32 + 8 pad (bf16)  -> pad codes (3,3): every 16 dw, 4 dw
#define GB_LD 136   // 128 + 8 pad (bf16) -> pad codes (5,3): every 64 dw, 4 dw

template <typename OutT>
__global__ __launch_bounds__(256) void gemm_bf16_kernel(
    const __bf16* __restrict__ A, const __bf16* __restrict__ Bm,
    OutT* __restrict__ C, int M, int N, int K) {
    __shared__ __bf16 As[2][128 * GA_LD];
    __shared__ __bf16 Bs[2][32 * GB_LD];

    const int tid  = threadIdx.x;
    const int lane = tid & 31;
    const int wave = tid >> 5;
    const int l15  = lane & 15;
    const int half = lane >> 4;
    const int wrow = (wave & 3) * 32;
    const int wcol = (wave >> 2) * 64;
    const int m0 = blockIdx.y * 128;
    const int n0 = blockIdx.x * 128;

    v8f acc[2][4];
#pragma unroll
    for (int t = 0; t < 2; ++t)
#pragma unroll
        for (int nt = 0; nt < 4; ++nt) acc[t][nt] = v8f_zero();

    auto stage = [&](int k0, int buf) {
        // A tile: 128 rows x 32 cols, row stride K; B tile: 32 rows x 128 cols
        tdm_load_2d(A + (size_t)m0 * K + k0, lds_offset_of(&As[buf][0]),
                    128u, 32u, (unsigned long long)K, 3u, 3u);
        tdm_load_2d(Bm + (size_t)k0 * N + n0, lds_offset_of(&Bs[buf][0]),
                    32u, 128u, (unsigned long long)N, 5u, 3u);
    };

    if (wave == 0) {
        stage(0, 0);
        __builtin_amdgcn_s_wait_tensorcnt(0);
    }
    __syncthreads();

    int p = 0;
    for (int k0 = 0; k0 < K; k0 += 32) {
        if (wave == 0 && k0 + 32 < K) stage(k0 + 32, p ^ 1);

        // A fragments: two 16x32 row tiles of this wave
        v16bf af[2];
#pragma unroll
        for (int t = 0; t < 2; ++t) {
            const __bf16* ap = &As[p][(wrow + t * 16 + l15) * GA_LD + half * 8];
#pragma unroll
            for (int i = 0; i < 8; ++i) af[t][i] = ap[i];
#pragma unroll
            for (int i = 0; i < 8; ++i) af[t][8 + i] = ap[16 + i];
        }
#pragma unroll
        for (int nt = 0; nt < 4; ++nt) {
            v16bf bfrag;
            const int col = wcol + nt * 16 + l15;
#pragma unroll
            for (int i = 0; i < 8; ++i)
                bfrag[i] = Bs[p][(half * 8 + i) * GB_LD + col];
#pragma unroll
            for (int i = 0; i < 8; ++i)
                bfrag[8 + i] = Bs[p][(half * 8 + 16 + i) * GB_LD + col];
            acc[0][nt] = __builtin_amdgcn_wmma_f32_16x16x32_bf16(
                false, af[0], false, bfrag, (short)0, acc[0][nt], false, false);
            acc[1][nt] = __builtin_amdgcn_wmma_f32_16x16x32_bf16(
                false, af[1], false, bfrag, (short)0, acc[1][nt], false, false);
        }

        if (wave == 0) __builtin_amdgcn_s_wait_tensorcnt(0);
        __syncthreads();
        p ^= 1;
    }

#pragma unroll
    for (int t = 0; t < 2; ++t)
#pragma unroll
        for (int nt = 0; nt < 4; ++nt)
#pragma unroll
            for (int r = 0; r < 8; ++r) {
                int row = m0 + wrow + t * 16 + r + half * 8;
                int col = n0 + wcol + nt * 16 + l15;
                C[(size_t)row * N + col] = (OutT)acc[t][nt][r];
            }
}

// ---------------------------------------------------------------------------
// In-place RoPE on bf16 tensor [BS, heads, 128]
// ---------------------------------------------------------------------------
__global__ void rope_kernel(__bf16* __restrict__ x, const int* __restrict__ pos,
                            int heads, size_t total) {
    size_t i = (size_t)blockIdx.x * blockDim.x + threadIdx.x;
    if (i >= total) return;
    int p = (int)(i & 63);
    size_t rest = i >> 6;
    int h = (int)(rest % heads);
    size_t bs = rest / heads;

    float pv  = (float)pos[bs];
    float inv = __powf(10000.f, -((float)(2 * p)) / 128.f);
    float f = pv * inv;
    float sn, cs;
    __sincosf(f, &sn, &cs);

    __bf16* base = x + (bs * (size_t)heads + h) * 128 + 2 * p;
    float xr = (float)base[0];
    float xi = (float)base[1];
    base[0] = (__bf16)(xr * cs - xi * sn);
    base[1] = (__bf16)(xr * sn + xi * cs);
}

// ---------------------------------------------------------------------------
// Flash attention: 4 waves / block, block covers 64 q rows of one (b,h).
// K/V 32-key chunks staged by TDM into LDS, shared by all 4 waves.
// ---------------------------------------------------------------------------
#define FA_LDP 40    // P tile row stride (32 + 8)
#define FA_LDV 136   // K/V tile row stride (128 + 8)

__global__ __launch_bounds__(128) void fa_kernel(
    const __bf16* __restrict__ Qb, const __bf16* __restrict__ Kb,
    const __bf16* __restrict__ Vb, const int* __restrict__ amask,
    const int* __restrict__ segid, __bf16* __restrict__ attnb, int S) {
    __shared__ __bf16 Ks[32 * FA_LDV];
    __shared__ __bf16 Vs[32 * FA_LDV];
    __shared__ __bf16 Ps[4][16 * FA_LDP];

    const int tid  = threadIdx.x;
    const int lane = tid & 31;
    const int wave = tid >> 5;
    const int l15  = lane & 15;
    const int half = lane >> 4;
    const int q0b  = blockIdx.x * 64;          // block q base
    const int qw   = q0b + wave * 16;          // this wave's q tile base
    const int h    = blockIdx.y;
    const int b    = blockIdx.z;
    const int kvh  = h >> 2;                   // GQA: H/KVH = 4
    const float scale = 0.08838834764831845f;  // 1/sqrt(128)

    // Q A-fragments: 4 head-dim slices of 32
    v16bf qa[4];
    {
        const int s = qw + l15;
        const __bf16* qrow = Qb + (((size_t)b * S + s) * 16 + h) * 128;
        const int kb = half * 8;
#pragma unroll
        for (int c = 0; c < 4; ++c) {
#pragma unroll
            for (int i = 0; i < 8; ++i) qa[c][i] = qrow[c * 32 + kb + i];
#pragma unroll
            for (int i = 0; i < 8; ++i) qa[c][8 + i] = qrow[c * 32 + kb + 16 + i];
        }
    }

    float rowM[8], rowL[8];
    int qidx[8], segq[8];
#pragma unroll
    for (int r = 0; r < 8; ++r) {
        rowM[r] = -3.0e38f;
        rowL[r] = 0.f;
        qidx[r] = qw + r + half * 8;
        segq[r] = segid[(size_t)b * S + qidx[r]];
    }
    v8f out[8];
#pragma unroll
    for (int t = 0; t < 8; ++t) out[t] = v8f_zero();

    const int nchunks = (q0b + 64 + 31) >> 5;
    const int kb2 = half * 8;
    for (int kc = 0; kc < nchunks; ++kc) {
        const int kbase = kc * 32;

        // TDM staging of K and V chunks (wave 0 drives the DMA)
        if (wave == 0) {
            const __bf16* kp = Kb + (((size_t)b * S + kbase) * 4 + kvh) * 128;
            const __bf16* vp = Vb + (((size_t)b * S + kbase) * 4 + kvh) * 128;
            tdm_load_2d(kp, lds_offset_of(&Ks[0]), 32u, 128u, 512ull, 5u, 3u);
            tdm_load_2d(vp, lds_offset_of(&Vs[0]), 32u, 128u, 512ull, 5u, 3u);
            __builtin_amdgcn_s_wait_tensorcnt(0);
        }
        __syncthreads();

        if (kbase <= qw + 15) {   // causal: chunk intersects this wave's rows
            // ---- scores = Q @ K^T (two 16-key halves) ----
            v8f sac[2];
            sac[0] = v8f_zero();
            sac[1] = v8f_zero();
#pragma unroll
            for (int kh = 0; kh < 2; ++kh) {
                const __bf16* krow = &Ks[(kh * 16 + l15) * FA_LDV];
#pragma unroll
                for (int c = 0; c < 4; ++c) {
                    v16bf bfrag;
#pragma unroll
                    for (int i = 0; i < 8; ++i) bfrag[i] = krow[c * 32 + kb2 + i];
#pragma unroll
                    for (int i = 0; i < 8; ++i) bfrag[8 + i] = krow[c * 32 + kb2 + 16 + i];
                    sac[kh] = __builtin_amdgcn_wmma_f32_16x16x32_bf16(
                        false, qa[c], false, bfrag, (short)0, sac[kh], false, false);
                }
            }

            float kbias[2];
            int kseg[2], kk[2];
#pragma unroll
            for (int kh = 0; kh < 2; ++kh) {
                const int key = kbase + kh * 16 + l15;
                kk[kh]    = key;
                kbias[kh] = (amask[(size_t)b * S + key] > 0) ? 0.f : NEGF;
                kseg[kh]  = segid[(size_t)b * S + key];
            }

            // ---- online softmax (rows per register across 16-lane halves) ----
#pragma unroll
            for (int r = 0; r < 8; ++r) {
                const int q = qidx[r];
                float s0 = sac[0][r] * scale + kbias[0];
                float s1 = sac[1][r] * scale + kbias[1];
                if (kk[0] > q || kseg[0] != segq[r]) s0 = NEGF;
                if (kk[1] > q || kseg[1] != segq[r]) s1 = NEGF;

                float cm = fmaxf(s0, s1);
#pragma unroll
                for (int off = 1; off < 16; off <<= 1)
                    cm = fmaxf(cm, __shfl_xor(cm, off, 32));
                const float newM  = fmaxf(rowM[r], cm);
                const float alpha = __expf(rowM[r] - newM);
                const float p0 = __expf(s0 - newM);
                const float p1 = __expf(s1 - newM);
                float rs = p0 + p1;
#pragma unroll
                for (int off = 1; off < 16; off <<= 1)
                    rs += __shfl_xor(rs, off, 32);
                rowL[r] = rowL[r] * alpha + rs;
                rowM[r] = newM;
#pragma unroll
                for (int t = 0; t < 8; ++t) out[t][r] = out[t][r] * alpha;

                const int m = r + half * 8;
                Ps[wave][m * FA_LDP + l15]      = (__bf16)p0;
                Ps[wave][m * FA_LDP + 16 + l15] = (__bf16)p1;
            }

            // ---- out += P @ V (P = one 16x32 A-fragment; same-wave DS order) ----
            v16bf pa;
            {
                const __bf16* prow = &Ps[wave][l15 * FA_LDP + half * 8];
#pragma unroll
                for (int i = 0; i < 8; ++i) pa[i] = prow[i];
#pragma unroll
                for (int i = 0; i < 8; ++i) pa[8 + i] = prow[16 + i];
            }
#pragma unroll
            for (int t = 0; t < 8; ++t) {
                v16bf vb;
#pragma unroll
                for (int i = 0; i < 8; ++i)
                    vb[i] = Vs[(kb2 + i) * FA_LDV + t * 16 + l15];
#pragma unroll
                for (int i = 0; i < 8; ++i)
                    vb[8 + i] = Vs[(kb2 + 16 + i) * FA_LDV + t * 16 + l15];
                out[t] = __builtin_amdgcn_wmma_f32_16x16x32_bf16(
                    false, pa, false, vb, (short)0, out[t], false, false);
            }
        }
        __syncthreads();
    }

    // ---- normalize and store bf16 for the Wo GEMM ----
#pragma unroll
    for (int r = 0; r < 8; ++r) {
        const float inv = 1.0f / rowL[r];
        const int q = qidx[r];
        __bf16* orow = attnb + (((size_t)b * S + q) * 16 + h) * 128;
#pragma unroll
        for (int t = 0; t < 8; ++t)
            orow[t * 16 + l15] = (__bf16)(out[t][r] * inv);
    }
}

// ---------------------------------------------------------------------------
// host launcher
// ---------------------------------------------------------------------------
extern "C" void kernel_launch(void* const* d_in, const int* in_sizes, int n_in,
                              void* d_out, int out_size, void* d_ws, size_t ws_size,
                              hipStream_t stream) {
    (void)in_sizes; (void)n_in; (void)out_size; (void)ws_size;
    constexpr int B = 2, S = 2048, D = 2048, H = 16, KVH = 4, HD = 128;
    constexpr int BS = B * S;                 // 4096
    constexpr int NQ = H * HD;                // 2048
    constexpr int NKV = KVH * HD;             // 512

    const float* hid  = (const float*)d_in[0];
    const int*   amsk = (const int*)d_in[1];
    const int*   pos  = (const int*)d_in[2];
    const int*   seg  = (const int*)d_in[3];
    const float* Wq   = (const float*)d_in[4];
    const float* Wk   = (const float*)d_in[5];
    const float* Wv   = (const float*)d_in[6];
    const float* Wo   = (const float*)d_in[7];
    float* outp = (float*)d_out;

    char* ws = (char*)d_ws;
    size_t off = 0;
    auto alloc = [&](size_t bytes) -> void* {
        void* p = ws + off;
        off = (off + bytes + 255) & ~(size_t)255;
        return p;
    };
    __bf16* Xb    = (__bf16*)alloc((size_t)BS * D * 2);
    __bf16* Wqb   = (__bf16*)alloc((size_t)D * NQ * 2);
    __bf16* Wkb   = (__bf16*)alloc((size_t)D * NKV * 2);
    __bf16* Wvb   = (__bf16*)alloc((size_t)D * NKV * 2);
    __bf16* Wob   = (__bf16*)alloc((size_t)NQ * D * 2);
    __bf16* Qb    = (__bf16*)alloc((size_t)BS * NQ * 2);
    __bf16* Kb    = (__bf16*)alloc((size_t)BS * NKV * 2);
    __bf16* Vb    = (__bf16*)alloc((size_t)BS * NKV * 2);
    __bf16* attnb = (__bf16*)alloc((size_t)BS * NQ * 2);

    auto cvt = [&](const float* src, __bf16* dst, size_t n) {
        f32_to_bf16_kernel<<<(unsigned)((n + 255) / 256), 256, 0, stream>>>(src, dst, n);
    };
    cvt(hid, Xb, (size_t)BS * D);
    cvt(Wq,  Wqb, (size_t)D * NQ);
    cvt(Wk,  Wkb, (size_t)D * NKV);
    cvt(Wv,  Wvb, (size_t)D * NKV);
    cvt(Wo,  Wob, (size_t)NQ * D);

    // QKV projections (bf16 out)
    gemm_bf16_kernel<__bf16><<<dim3(NQ / 128, BS / 128), 256, 0, stream>>>(
        Xb, Wqb, Qb, BS, NQ, D);
    gemm_bf16_kernel<__bf16><<<dim3(NKV / 128, BS / 128), 256, 0, stream>>>(
        Xb, Wkb, Kb, BS, NKV, D);
    gemm_bf16_kernel<__bf16><<<dim3(NKV / 128, BS / 128), 256, 0, stream>>>(
        Xb, Wvb, Vb, BS, NKV, D);

    // RoPE on Q and K (in place)
    {
        size_t tq = (size_t)BS * H * 64;
        rope_kernel<<<(unsigned)((tq + 255) / 256), 256, 0, stream>>>(Qb, pos, H, tq);
        size_t tk = (size_t)BS * KVH * 64;
        rope_kernel<<<(unsigned)((tk + 255) / 256), 256, 0, stream>>>(Kb, pos, KVH, tk);
    }

    // flash attention
    fa_kernel<<<dim3(S / 64, H, B), 128, 0, stream>>>(Qb, Kb, Vb, amsk, seg, attnb, S);

    // output projection (fp32 out)
    gemm_bf16_kernel<float><<<dim3(D / 128, BS / 128), 256, 0, stream>>>(
        attnb, Wob, outp, BS, D, NQ);
}
